// LlamaMoEAttention_15307263443368
// MI455X (gfx1250) — matrix operand order repaired and checked by reading, not compiled, not run on hardware
//
#include <hip/hip_runtime.h>

// ---------------------------------------------------------------------------
// Types for CDNA5 WMMA (wave32): v_wmma_f32_16x16x32_bf16
// ---------------------------------------------------------------------------
typedef __bf16 bf16_t;
typedef __attribute__((ext_vector_type(16))) __bf16 v16bf;
typedef __attribute__((ext_vector_type(8)))  __bf16 v8bf;
typedef __attribute__((ext_vector_type(8)))  float  v8f;

union ABf { v16bf v; v8bf h[2]; };

#define S_LEN 4096
#define HID   2048
#define NH    16
#define NKV   4
#define HD    128

// The TDM writes LDS behind the compiler's back (the intrinsic has no pointer
// operand), so alias analysis would otherwise fold the subsequent ds_loads of
// "never-written" LDS to undef. This zero-instruction barrier escapes the LDS
// pointer and clobbers memory so the compiler must assume LDS was modified.
#define LDS_PUBLISH(p) asm volatile("" :: "r"((void*)(p)) : "memory")

// ---------------------------------------------------------------------------
// Tensor Data Mover support (gfx1250). Guarded so compilation never breaks.
// This toolchain exposes the 6-arg builtin:
//   (uint32x4 g0, int32x8 g1, int32x4 g2, int32x4 g3, int32x8 extra, i32 cpol)
// ---------------------------------------------------------------------------
#if defined(__AMDGCN__) && __has_builtin(__builtin_amdgcn_tensor_load_to_lds) && \
    __has_builtin(__builtin_amdgcn_s_wait_tensorcnt)
#define HAVE_TDM 1
#else
#define HAVE_TDM 0
#endif

#if HAVE_TDM
typedef __attribute__((ext_vector_type(4))) unsigned int u32x4;
typedef __attribute__((ext_vector_type(8))) int          i32x8;
typedef __attribute__((ext_vector_type(4))) int          i32x4;

// Issue a TDM load of a 2D tile (tile_d0 contiguous elems x tile_d1 rows,
// bf16, row stride = stride0 elems) from global into LDS at byte offset
// lds_byte. D# packing per CDNA5 ISA 8.3/8.4:
//  g0: count=1 | lds_addr<<32 | global_addr<<64 | type=2 @126
//  g1: data_size=1(2B)@16, tensor_dim0/1 = 2^30 (never OOB here),
//      tile_dim0@112, tile_dim1@128, tensor_dim0_stride@160
__device__ __forceinline__ void tdm_load_2d(unsigned lds_byte, const void* gp,
                                            unsigned tile_d0, unsigned tile_d1,
                                            unsigned long long stride0)
{
    unsigned long long ga = (unsigned long long)gp;
    const unsigned D0 = 0x40000000u, D1 = 0x40000000u;
    u32x4 g0 = { 1u,                                   // count=1 (valid D#)
                 lds_byte,                             // lds_addr
                 (unsigned)(ga & 0xffffffffu),         // global_addr[31:0]
                 (unsigned)((ga >> 32) & 0x1ffffffu)   // global_addr[56:32]
                     | 0x80000000u };                  // type=2
    i32x8 g1 = { (int)0x00010000u,                             // data_size=2B
                 (int)((D0 & 0xffffu) << 16),                  // dim0[15:0]
                 (int)(((D0 >> 16) & 0xffffu) | ((D1 & 0xffffu) << 16)),
                 (int)(((D1 >> 16) & 0xffffu) | (tile_d0 << 16)),
                 (int)(tile_d1 & 0xffffu),                     // tile_dim1
                 (int)(stride0 & 0xffffffffu),                 // stride0 lo
                 (int)((stride0 >> 32) & 0xffffu),             // stride0 hi
                 0 };
    i32x4 z4 = { 0, 0, 0, 0 };
    i32x8 z8 = { 0, 0, 0, 0, 0, 0, 0, 0 };
    __builtin_amdgcn_tensor_load_to_lds(g0, g1, z4, z4, z8, 0);
}
#endif

// ---------------------------------------------------------------------------
// Kernel 0: fp32 -> bf16 conversion (grid-stride)
// ---------------------------------------------------------------------------
__global__ void cvt_f32_bf16(const float* __restrict__ src,
                             bf16_t* __restrict__ dst, size_t n)
{
    size_t i = (size_t)blockIdx.x * blockDim.x + threadIdx.x;
    size_t stride = (size_t)gridDim.x * blockDim.x;
    for (; i < n; i += stride) dst[i] = (bf16_t)src[i];
}

// ---------------------------------------------------------------------------
// Kernel 1: NT GEMM  C[M,N] = A[M,K] * B[N,K]^T  (X @ W.T), bf16 in,
// bf16 or fp32 out. Block = 256 threads = 8 waves; block tile 128(M)x64(N),
// per-wave 16x64. Operand k-slices (A 128x32, B 64x32) are staged into
// double-buffered LDS by the Tensor Data Mover (wave 0 issues descriptors,
// TENSORcnt + barrier synchronize), so global latency hides behind WMMAs
// and the 8 waves share one copy of B instead of re-fetching it 8x.
// LDS element map: [A buf0 @0][A buf1 @4096][B buf0 @8192][B buf1 @10240]
// ---------------------------------------------------------------------------
__global__ void __launch_bounds__(256)
gemm_nt_kernel(const bf16_t* __restrict__ A, const bf16_t* __restrict__ B,
               bf16_t* __restrict__ Cb, float* __restrict__ Cf,
               int M, int N, int K, int store_f32)
{
    // Single LDS object => sits at LDS offset 0 (assumed by TDM descriptors).
    __shared__ __align__(32) bf16_t smem[2 * 4096 + 2 * 2048];

    const int tid  = threadIdx.x;
    const int lane = tid & 31;
    const int wave = tid >> 5;
    const int l16  = lane & 15;
    const int hi   = lane >> 4;
    const int m0   = blockIdx.y * 128;
    const int n0   = blockIdx.x * 64;

    auto stage = [&](int k0, int buf) {
#if HAVE_TDM
        if (wave == 0) {
            // byte offsets of the LDS destination buffers
            tdm_load_2d((unsigned)(buf * 8192), A + (size_t)m0 * K + k0,
                        32u, 128u, (unsigned long long)K);
            tdm_load_2d((unsigned)(16384 + buf * 4096), B + (size_t)n0 * K + k0,
                        32u, 64u, (unsigned long long)K);
        }
#else
        {   // cooperative fallback: 256 threads move A(8KB) + B(4KB)
            bf16_t* dA = smem + buf * 4096;
            bf16_t* dB = smem + 8192 + buf * 2048;
            int row = tid >> 1, half = tid & 1;
            *(v16bf*)(dA + row * 32 + half * 16) =
                *(const v16bf*)(A + (size_t)(m0 + row) * K + k0 + half * 16);
            if (tid < 128) {
                int rb = tid >> 1, hb = tid & 1;
                *(v16bf*)(dB + rb * 32 + hb * 16) =
                    *(const v16bf*)(B + (size_t)(n0 + rb) * K + k0 + hb * 16);
            }
        }
#endif
    };

    stage(0, 0);
#if HAVE_TDM
    if (wave == 0) __builtin_amdgcn_s_wait_tensorcnt(0);
#endif
    LDS_PUBLISH(smem);        // TDM wrote LDS: tell the optimizer
    __syncthreads();

    v8f acc[4] = {};
    const int msub = wave * 16;

    for (int k0 = 0; k0 < K; k0 += 32) {
        const int buf = (k0 >> 5) & 1;
        if (k0 + 32 < K) stage(k0 + 32, buf ^ 1);  // async prefetch next slice

        const bf16_t* sAbuf = smem + buf * 4096;
        const bf16_t* sBbuf = smem + 8192 + buf * 2048;

        ABf a;
        a.v = *(const v16bf*)(sAbuf + (msub + l16) * 32 + hi * 16);
#pragma unroll
        for (int t = 0; t < 4; ++t) {
            v16bf b = *(const v16bf*)(sBbuf + (t * 16 + l16) * 32 + hi * 16);
            acc[t] = __builtin_amdgcn_wmma_f32_16x16x32_bf16(
                false, a.v, false, b, (short)0, acc[t], false, false);
        }

#if HAVE_TDM
        if (wave == 0) __builtin_amdgcn_s_wait_tensorcnt(0);
#endif
        LDS_PUBLISH(smem);    // next slice resident in LDS
        __syncthreads();      // everyone done with 'buf'
    }

#pragma unroll
    for (int t = 0; t < 4; ++t) {
#pragma unroll
        for (int r = 0; r < 8; ++r) {
            int row = m0 + msub + r + hi * 8;
            int col = n0 + t * 16 + l16;
            float v = acc[t][r];
            if (store_f32) Cf[(size_t)row * N + col] = v;
            else           Cb[(size_t)row * N + col] = (bf16_t)v;
        }
    }
}

// ---------------------------------------------------------------------------
// Kernel 2: RoPE in-place on bf16 Q [S,16,128] and K [S,4,128]
// ---------------------------------------------------------------------------
__global__ void rope_kernel(bf16_t* __restrict__ Qb, bf16_t* __restrict__ Kb,
                            const int* __restrict__ pos)
{
    size_t idx = (size_t)blockIdx.x * blockDim.x + threadIdx.x;
    const size_t total = (size_t)S_LEN * (NH + NKV) * 64;
    if (idx >= total) return;
    int d    = (int)(idx & 63);
    size_t t = idx >> 6;
    int h    = (int)(t % (NH + NKV));
    int s    = (int)(t / (NH + NKV));

    float p        = (float)pos[s];
    float inv_freq = __expf(-((float)(2 * d) / (float)HD) * __logf(10000.0f));
    float ang      = p * inv_freq;
    float c = __cosf(ang), si = __sinf(ang);

    bf16_t* base = (h < NH) ? (Qb + (size_t)s * HID + h * HD)
                            : (Kb + (size_t)s * (NKV * HD) + (h - NH) * HD);
    float x0 = (float)base[d];
    float x1 = (float)base[d + 64];
    base[d]      = (bf16_t)(x0 * c - x1 * si);
    base[d + 64] = (bf16_t)(x1 * c + x0 * si);
}

// ---------------------------------------------------------------------------
// Kernel 3: causal flash attention, GQA 16Q/4KV heads, head_dim 128.
// One wave = 16 query rows of one head; 32-key causal blocks; online softmax
// with width-16 shfl_xor butterflies; P re-laid-out C->A through LDS; V
// staged transposed in LDS so PV B-fragments are contiguous ds_load_b128s.
// ---------------------------------------------------------------------------
#define VSTRIDE  48
#define FA_WAVES 4
#define W_LDS    (HD * VSTRIDE + 16 * 32)

__global__ void __launch_bounds__(128)
flash_attn_kernel(const bf16_t* __restrict__ Qb, const bf16_t* __restrict__ Kb,
                  const bf16_t* __restrict__ Vb, bf16_t* __restrict__ Ob)
{
    __shared__ __align__(32) bf16_t smem[FA_WAVES * W_LDS];

    const int lane = threadIdx.x & 31;
    const int wave = threadIdx.x >> 5;
    const int l16  = lane & 15;
    const int hi   = lane >> 4;

    const int gwave = blockIdx.x * FA_WAVES + wave;
    const int head  = gwave >> 8;
    const int qblk  = gwave & 255;
    const int kvh   = head >> 2;

    bf16_t* vT = smem + wave * W_LDS;
    bf16_t* pT = vT + HD * VSTRIDE;

    ABf qa[4];
    {
        const bf16_t* qr = Qb + (size_t)(qblk * 16 + l16) * HID + head * HD;
#pragma unroll
        for (int kk = 0; kk < 4; ++kk) {
            qa[kk].v = *(const v16bf*)(qr + kk * 32 + hi * 16);
        }
    }

    v8f acc[8] = {};
    float mrow[8], lrow[8];
#pragma unroll
    for (int r = 0; r < 8; ++r) { mrow[r] = -3.0e38f; lrow[r] = 0.0f; }

    const float scale = 0.08838834764831845f;   // 1/sqrt(128)

    for (int j = 0; j < (qblk + 1) * 16; j += 32) {
        // -------- stage V^T (32 keys x 128 dims) into LDS, packed b32 stores
        {
            int k0 = j + l16 * 2; if (k0 >= S_LEN) k0 = S_LEN - 1;
            int k1 = k0 + 1;      if (k1 >= S_LEN) k1 = S_LEN - 1;
            const bf16_t* v0p = Vb + (size_t)k0 * (NKV * HD) + kvh * HD + hi * 64;
            const bf16_t* v1p = Vb + (size_t)k1 * (NKV * HD) + kvh * HD + hi * 64;
            __builtin_prefetch(v0p + 32 * (NKV * HD), 0, 3);   // next key block
#pragma unroll
            for (int d = 0; d < 64; d += 8) {
                v8bf r0 = *(const v8bf*)(v0p + d);
                v8bf r1 = *(const v8bf*)(v1p + d);
#pragma unroll
                for (int e = 0; e < 8; ++e) {
                    __bf16 b0 = r0[e], b1 = r1[e];
                    unsigned int wv =
                        (unsigned int)__builtin_bit_cast(unsigned short, b0) |
                        ((unsigned int)__builtin_bit_cast(unsigned short, b1) << 16);
                    *(unsigned int*)(vT + (size_t)(hi * 64 + d + e) * VSTRIDE
                                        + l16 * 2) = wv;
                }
            }
        }

        // -------- scores: two 16x16 tiles of Q*K^T over this 32-key block
        v8f s0 = {}, s1 = {};
        {
            int key0 = j + l16;      if (key0 >= S_LEN) key0 = S_LEN - 1;
            int key1 = j + 16 + l16; if (key1 >= S_LEN) key1 = S_LEN - 1;
            const bf16_t* kr0 = Kb + (size_t)key0 * (NKV * HD) + kvh * HD + hi * 16;
            const bf16_t* kr1 = Kb + (size_t)key1 * (NKV * HD) + kvh * HD + hi * 16;
            __builtin_prefetch(kr0 + 32 * (NKV * HD), 0, 3);
#pragma unroll
            for (int kk = 0; kk < 4; ++kk) {
                v16bf b0 = *(const v16bf*)(kr0 + kk * 32);
                v16bf b1 = *(const v16bf*)(kr1 + kk * 32);
                s0 = __builtin_amdgcn_wmma_f32_16x16x32_bf16(
                    false, qa[kk].v, false, b0, (short)0, s0, false, false);
                s1 = __builtin_amdgcn_wmma_f32_16x16x32_bf16(
                    false, qa[kk].v, false, b1, (short)0, s1, false, false);
            }
        }

        // -------- online softmax (row = r + 8*hi, cols striped over 16 lanes)
#pragma unroll
        for (int r = 0; r < 8; ++r) {
            int m    = r + hi * 8;
            int qidx = qblk * 16 + m;
            float v0 = s0[r] * scale;
            float v1 = s1[r] * scale;
            if (j + l16 > qidx)      v0 = -3.0e38f;
            if (j + 16 + l16 > qidx) v1 = -3.0e38f;
            float mx = fmaxf(v0, v1);
            mx = fmaxf(mx, __shfl_xor(mx, 8, 16));
            mx = fmaxf(mx, __shfl_xor(mx, 4, 16));
            mx = fmaxf(mx, __shfl_xor(mx, 2, 16));
            mx = fmaxf(mx, __shfl_xor(mx, 1, 16));
            float mnew = fmaxf(mrow[r], mx);
            float al   = __expf(mrow[r] - mnew);
            float p0   = __expf(v0 - mnew);
            float p1   = __expf(v1 - mnew);
            float rs = p0 + p1;
            rs += __shfl_xor(rs, 8, 16);
            rs += __shfl_xor(rs, 4, 16);
            rs += __shfl_xor(rs, 2, 16);
            rs += __shfl_xor(rs, 1, 16);
            lrow[r] = lrow[r] * al + rs;
            mrow[r] = mnew;
#pragma unroll
            for (int t = 0; t < 8; ++t) acc[t][r] *= al;
            pT[m * 32 + l16]      = (bf16_t)p0;
            pT[m * 32 + 16 + l16] = (bf16_t)p1;
        }

        asm volatile("s_wait_dscnt 0x0" ::: "memory");

        // -------- PV: acc(16x128) += P(16x32) * V(32x128), 8 dim tiles
        ABf pa;
        pa.v = *(const v16bf*)(pT + l16 * 32 + hi * 16);
#pragma unroll
        for (int t = 0; t < 8; ++t) {
            const bf16_t* vr = vT + (size_t)(t * 16 + l16) * VSTRIDE + hi * 16;
            v16bf bv = *(const v16bf*)vr;
            acc[t] = __builtin_amdgcn_wmma_f32_16x16x32_bf16(
                false, pa.v, false, bv, (short)0, acc[t], false, false);
        }
    }

#pragma unroll
    for (int t = 0; t < 8; ++t) {
#pragma unroll
        for (int r = 0; r < 8; ++r) {
            int m = r + hi * 8;
            float o = acc[t][r] / lrow[r];
            Ob[(size_t)(qblk * 16 + m) * HID + head * HD + t * 16 + l16] = (bf16_t)o;
        }
    }
}

// ---------------------------------------------------------------------------
// Host-side orchestration
// ---------------------------------------------------------------------------
extern "C" void kernel_launch(void* const* d_in, const int* in_sizes, int n_in,
                              void* d_out, int out_size, void* d_ws, size_t ws_size,
                              hipStream_t stream)
{
    (void)in_sizes; (void)n_in; (void)out_size; (void)ws_size;
    const float* hs  = (const float*)d_in[0];   // [1,4096,2048]
    const float* Wq  = (const float*)d_in[1];   // [2048,2048]
    const float* Wk  = (const float*)d_in[2];   // [512,2048]
    const float* Wv  = (const float*)d_in[3];   // [512,2048]
    const float* Wo  = (const float*)d_in[4];   // [2048,2048]
    const int*   pos = (const int*)d_in[5];     // [1,4096]
    float*       out = (float*)d_out;           // [1,4096,2048]

    // Workspace layout (bf16), total ~76 MB
    bf16_t* Xb  = (bf16_t*)d_ws;
    bf16_t* Wqb = Xb  + (size_t)S_LEN * HID;
    bf16_t* Wkb = Wqb + (size_t)HID * HID;
    bf16_t* Wvb = Wkb + (size_t)(NKV * HD) * HID;
    bf16_t* Wob = Wvb + (size_t)(NKV * HD) * HID;
    bf16_t* Qb  = Wob + (size_t)HID * HID;
    bf16_t* Kb  = Qb  + (size_t)S_LEN * HID;
    bf16_t* Vb  = Kb  + (size_t)S_LEN * (NKV * HD);
    bf16_t* Ab  = Vb  + (size_t)S_LEN * (NKV * HD);

    cvt_f32_bf16<<<2048, 256, 0, stream>>>(hs, Xb, (size_t)S_LEN * HID);
    cvt_f32_bf16<<<2048, 256, 0, stream>>>(Wq, Wqb, (size_t)HID * HID);
    cvt_f32_bf16<<<512,  256, 0, stream>>>(Wk, Wkb, (size_t)(NKV * HD) * HID);
    cvt_f32_bf16<<<512,  256, 0, stream>>>(Wv, Wvb, (size_t)(NKV * HD) * HID);
    cvt_f32_bf16<<<2048, 256, 0, stream>>>(Wo, Wob, (size_t)HID * HID);

    gemm_nt_kernel<<<dim3(HID / 64, S_LEN / 128), 256, 0, stream>>>(
        Xb, Wqb, Qb, (float*)nullptr, S_LEN, HID, HID, 0);
    gemm_nt_kernel<<<dim3((NKV * HD) / 64, S_LEN / 128), 256, 0, stream>>>(
        Xb, Wkb, Kb, (float*)nullptr, S_LEN, NKV * HD, HID, 0);
    gemm_nt_kernel<<<dim3((NKV * HD) / 64, S_LEN / 128), 256, 0, stream>>>(
        Xb, Wvb, Vb, (float*)nullptr, S_LEN, NKV * HD, HID, 0);

    {
        size_t n = (size_t)S_LEN * (NH + NKV) * 64;
        rope_kernel<<<(unsigned)((n + 255) / 256), 256, 0, stream>>>(Qb, Kb, pos);
    }

    flash_attn_kernel<<<(S_LEN / 16) * NH / FA_WAVES, 128, 0, stream>>>(
        Qb, Kb, Vb, Ab);

    gemm_nt_kernel<<<dim3(HID / 64, S_LEN / 128), 256, 0, stream>>>(
        Ab, Wob, (bf16_t*)nullptr, out, S_LEN, HID, HID, 1);
}
